// SSR2_44032004718756
// MI455X (gfx1250) — matrix operand order repaired and checked by reading, not compiled, not run on hardware
//
#include <hip/hip_runtime.h>
#include <hip/hip_bf16.h>
#include <cstdint>

// Model dims (compile-time)
#define V_   96
#define D_   256
#define NS_  24
#define H_   512
#define L_   2
#define B_   64
#define T_   128
#define D3_  768   // 3*D

typedef __attribute__((ext_vector_type(16))) _Float16 v16h;
typedef __attribute__((ext_vector_type(8)))  _Float16 v8h;
typedef __attribute__((ext_vector_type(8)))  float    v8f;
typedef __attribute__((ext_vector_type(4)))  uint32_t u32x4;
typedef __attribute__((ext_vector_type(8)))  uint32_t u32x8;

// ---------------------------------------------------------------- utilities
__device__ __forceinline__ float gelu_exact(float x) {
    return 0.5f * x * (1.0f + erff(x * 0.70710678118654752f));
}
__device__ __forceinline__ float sigm(float x) { return 1.0f / (1.0f + __expf(-x)); }

// block-wide sum, blockDim.x == 256, red must be 256 floats of LDS
__device__ __forceinline__ float block_sum256(float v, float* red) {
    int tid = threadIdx.x;
    red[tid] = v; __syncthreads();
    #pragma unroll
    for (int s = 128; s > 0; s >>= 1) {
        if (tid < s) red[tid] += red[tid + s];
        __syncthreads();
    }
    float r = red[0]; __syncthreads();
    return r;
}

// ---------------------------------------------------------------- setup kernels
__global__ void k_zero(float* p, int n) {
    int i = blockIdx.x * 256 + threadIdx.x;
    if (i < n) p[i] = 0.0f;
}
__global__ void k_cvt_f16(const float* __restrict__ src, _Float16* __restrict__ dst, int n) {
    int i = blockIdx.x * 256 + threadIdx.x;
    if (i < n) dst[i] = (_Float16)src[i];
}

// xv[b,d] = emb[x[b,t], d] + pos_emb[t, d]
__global__ void k_embed(const int* __restrict__ x, const float* __restrict__ emb,
                        const float* __restrict__ pos, float* __restrict__ xv, int t) {
    int i = blockIdx.x * 256 + threadIdx.x;       // i in [0, B*D)
    int b = i >> 8, d = i & 255;
    int tok = x[b * T_ + t];
    xv[i] = emb[tok * D_ + d] + pos[t * D_ + d];
}

// ---------------------------------------------------------------- routing
// One block per batch row b. Computes slot_mean, gelu MLP, softmax -> alpha[b,:]
__global__ void k_route(const float* __restrict__ S,      // [B*NS, D] (layer slice)
                        const float* __restrict__ xv,     // [B, D]
                        const float* __restrict__ w1,     // [2D, D]
                        const float* __restrict__ b1,     // [D]
                        const float* __restrict__ w2,     // [D, NS]
                        const float* __restrict__ b2,     // [NS]
                        float* __restrict__ alpha) {      // [B, NS]
    __shared__ float ctx[2 * D_];
    __shared__ float hm[D_];
    __shared__ float lg[NS_];
    int b = blockIdx.x, d = threadIdx.x;

    float sm = 0.0f;
    #pragma unroll 4
    for (int s = 0; s < NS_; ++s) sm += S[(b * NS_ + s) * D_ + d];
    ctx[d]      = xv[b * D_ + d];
    ctx[D_ + d] = sm * (1.0f / NS_);
    __syncthreads();

    float acc = b1[d];
    for (int k = 0; k < 2 * D_; ++k) acc += ctx[k] * w1[k * D_ + d];
    hm[d] = gelu_exact(acc);
    __syncthreads();

    if (d < NS_) {
        float a = b2[d];
        for (int k = 0; k < D_; ++k) a += hm[k] * w2[k * NS_ + d];
        lg[d] = a;
    }
    __syncthreads();
    if (d == 0) {   // tiny serial softmax over 24 entries (TAU = 1)
        float mx = lg[0];
        for (int s = 1; s < NS_; ++s) mx = fmaxf(mx, lg[s]);
        float ssum = 0.0f;
        float e[NS_];
        for (int s = 0; s < NS_; ++s) { e[s] = __expf(lg[s] - mx); ssum += e[s]; }
        float inv = 1.0f / ssum;
        for (int s = 0; s < NS_; ++s) alpha[b * NS_ + s] = e[s] * inv;
    }
}

// Xw[b,e] = sum_k xv[b,k] * wih[e,k]   (no bias; bih folded into GRU epilogue)
__global__ void k_gx(const float* __restrict__ xv, const float* __restrict__ wih,
                     float* __restrict__ Xw) {
    __shared__ float xr[D_];
    int b = blockIdx.x, tid = threadIdx.x;
    xr[tid] = xv[b * D_ + tid];
    __syncthreads();
    #pragma unroll
    for (int g = 0; g < 3; ++g) {
        int e = g * D_ + tid;
        const float* w = wih + e * D_;
        float a = 0.0f;
        for (int k = 0; k < D_; ++k) a += xr[k] * w[k];
        Xw[b * D3_ + e] = a;
    }
}

// ---------------------------------------------------------------- GRU: WMMA GEMM + fused gates
// gh = S @ Whh^T over columns {d, D+d, 2D+d}; epilogue forms r,z,n and S_new.
// Grid: 24 M-tiles (64 rows) x 8 D-tiles (32 cols) = 192 blocks, 128 threads (4 waves).
// Whh tile (96 cols x K=256, f16, 48KB) DMA'd once per workgroup into LDS by the
// Tensor Data Mover (tensor_load_to_lds, 3-D tile: X=K, Y=32 d-rows, Z=3 gates);
// WMMA B fragments then come from LDS (ds_load_b128, conflict-free 32B runs).
__global__ void k_gru(const float*  __restrict__ Scur,   // [B*NS, D] layer slice (read)
                      float*        __restrict__ Snxt,   // [B*NS, D] layer slice (write)
                      const _Float16* __restrict__ whh16,// [3D, D] f16 weights
                      const float*  __restrict__ alpha,  // [B, NS]
                      const float*  __restrict__ Xw,     // [B, 3D]
                      const float*  __restrict__ bih,    // [3D]
                      const float*  __restrict__ bhh) {  // [3D]
    __shared__ __align__(32) _Float16 Bs[96 * D_];       // [c=g*32+dloc][k], 48KB

    const int mt   = blockIdx.x % 24;
    const int dt   = blockIdx.x / 24;
    const int tid  = threadIdx.x;
    const int wave = tid >> 5;
    const int lane = tid & 31;
    const int row0 = mt * 64 + wave * 16;
    const int d0   = dt * 32;

    // ---- TDM fill: one wave issues a 3-D tensor_load_to_lds for the whole tile.
    // D# per CDNA5 ISA: tile X=256 halves (contiguous),
    // Y=32 rows (stride 256 elems), Z=3 gates (stride 65536 elems). The TDM's
    // linear LDS fill order (X,Y,Z) matches Bs[(g*32+y)*256 + x] exactly.
    if (tid < 32) {
        uint32_t lds_base = __builtin_amdgcn_readfirstlane(
            (uint32_t)(uintptr_t)(void*)&Bs[0]);            // LDS aperture: addr[31:0] = LDS offset
        uint64_t ga = (uint64_t)(uintptr_t)(whh16 + (size_t)d0 * D_);
        uint32_t ga_lo = __builtin_amdgcn_readfirstlane((uint32_t)(ga & 0xFFFFFFFFull));
        uint32_t ga_hi = __builtin_amdgcn_readfirstlane((uint32_t)(ga >> 32));

        u32x4 g0;                       // D# group 0 (128b)
        g0[0] = 1u;                                  // count=1, user descriptor
        g0[1] = lds_base;                            // lds_addr [63:32]
        g0[2] = ga_lo;                               // global_addr [95:64]
        g0[3] = (ga_hi & 0x01FFFFFFu) | (2u << 30);  // global_addr[56:32] | type=2

        u32x8 g1;                       // D# group 1 (256b)
        g1[0] = 0x00010000u;   // workgroup_mask=0 | data_size=1 (2B)
        g1[1] = 0x01000000u;   // atomic_barrier_addr=0 | tensor_dim0 lo16 = 256
        g1[2] = 0x01000000u;   // tensor_dim0 hi16 = 0  | tensor_dim1 lo16 = 256
        g1[3] = 0x01000000u;   // tensor_dim1 hi16 = 0  | tile_dim0 = 256
        g1[4] = 0x00030020u;   // tile_dim1 = 32        | tile_dim2 = 3
        g1[5] = 256u;          // tensor_dim0_stride [31:0] = 256
        g1[6] = 0u;            // dim0_stride hi16 = 0  | tensor_dim1_stride lo16 = 0
        g1[7] = 1u;            // tensor_dim1_stride [47:16] = 1 -> stride 65536

        u32x4 g2;                       // D# group 2 (128b)
        g2[0] = 3u;            // tensor_dim2 = 3
        g2[1] = 0u;            // tensor_dim3 (unused)
        g2[2] = 65536u;        // tensor_dim2_stride [31:0]
        g2[3] = 0u;            // dim2_stride hi | tile_dim3 = 0

        u32x4 g3;                       // D# group 3 (128b): unused dims
        g3[0] = 0u; g3[1] = 0u; g3[2] = 0u; g3[3] = 0u;

        asm volatile("tensor_load_to_lds %0, %1, %2, %3"
                     :
                     : "s"(g0), "s"(g1), "s"(g2), "s"(g3)
                     : "memory");
        __builtin_amdgcn_s_wait_tensorcnt(0);        // TENSORcnt == 0: tile in LDS
    }
    __syncthreads();

    v8f acc[6] = {};   // acc[g*2+h]

    const int mA    = lane & 15;        // A-matrix row for this lane
    const int halfA = lane >> 4;        // A-matrix K-half select
    const int nB    = lane & 15;        // B-matrix column for this lane
    const int halfB = lane >> 4;        // B-matrix K-half select

    for (int k0 = 0; k0 < D_; k0 += 32) {
        // --- A fragment: 16x32 f16 tile of S (contiguous k = 16*halfA + s) -------------
        v16h A;
        const float* arow = Scur + (row0 + mA) * D_ + k0;
        #pragma unroll
        for (int v = 0; v < 8; ++v) {
            int kb = 2 * v + 16 * halfA;   // K pair base
            A[2 * v]     = (_Float16)arow[kb];
            A[2 * v + 1] = (_Float16)arow[kb + 1];
        }
        // --- B fragments from LDS + WMMA ---------------------------------------
        #pragma unroll
        for (int g = 0; g < 3; ++g) {
            #pragma unroll
            for (int h = 0; h < 2; ++h) {
                int c = g * 32 + h * 16 + nB;                  // staged column index
                const v8h* bp = (const v8h*)(Bs + c * D_ + k0 + 16 * halfB);
                v8h blo = bp[0], bhi = bp[1];                  // 2x ds_load_b128
                v16h Bt;
                #pragma unroll
                for (int q = 0; q < 8; ++q) { Bt[q] = blo[q]; Bt[8 + q] = bhi[q]; }
                acc[g * 2 + h] = __builtin_amdgcn_wmma_f32_16x16x32_f16(
                    false, A, false, Bt, (short)0, acc[g * 2 + h], false, false);
            }
        }
    }

    // --- fused GRU gate epilogue: C layout VGPR j, M = j + 8*(lane>>4), N = lane&15
    const int mhi = (lane >> 4) * 8;
    const int nn  = lane & 15;
    #pragma unroll
    for (int h = 0; h < 2; ++h) {
        int d = d0 + h * 16 + nn;
        float bhr = bhh[d], bhz = bhh[D_ + d], bhn = bhh[2 * D_ + d];
        float bir = bih[d], biz = bih[D_ + d], bin = bih[2 * D_ + d];
        #pragma unroll
        for (int j = 0; j < 8; ++j) {
            int row = row0 + j + mhi;
            int b = row / NS_, s = row % NS_;
            float al = alpha[b * NS_ + s];
            float hr = acc[0 + h][j] + bhr;
            float hz = acc[2 + h][j] + bhz;
            float hn = acc[4 + h][j] + bhn;
            float ir = al * Xw[b * D3_ + d]           + bir;
            float iz = al * Xw[b * D3_ + D_ + d]      + biz;
            float in = al * Xw[b * D3_ + 2 * D_ + d]  + bin;
            float r  = sigm(ir + hr);
            float z  = sigm(iz + hz);
            float n  = tanhf(in + r * hn);
            float so = Scur[row * D_ + d];
            Snxt[row * D_ + d] = (1.0f - z) * n + z * so;
        }
    }
}

// ---------------------------------------------------------------- slot ctx + proj + LN1
__global__ void k_slotproj(const float* __restrict__ alpha, const float* __restrict__ Snew,
                           const float* __restrict__ xv,
                           const float* __restrict__ pw, const float* __restrict__ pb,
                           const float* __restrict__ g1, const float* __restrict__ be1,
                           float* __restrict__ x_mid, float* __restrict__ h1) {
    __shared__ float al[NS_];
    __shared__ float sc[D_];
    __shared__ float red[256];
    int b = blockIdx.x, d = threadIdx.x;
    if (d < NS_) al[d] = alpha[b * NS_ + d];
    __syncthreads();

    float v = 0.0f;
    #pragma unroll 4
    for (int s = 0; s < NS_; ++s) v += al[s] * Snew[(b * NS_ + s) * D_ + d];
    sc[d] = v;
    __syncthreads();

    float xm = xv[b * D_ + d] + pb[d];
    for (int k = 0; k < D_; ++k) xm += sc[k] * pw[k * D_ + d];
    x_mid[b * D_ + d] = xm;

    float mean = block_sum256(xm, red) * (1.0f / D_);
    float c = xm - mean;
    float var = block_sum256(c * c, red) * (1.0f / D_);
    float rs = rsqrtf(var + 1e-5f);
    h1[b * D_ + d] = c * rs * g1[d] + be1[d];
}

// ---------------------------------------------------------------- FFN + LN2 -> xv (in place)
__global__ void k_ffn(const float* __restrict__ x_mid, const float* __restrict__ h1,
                      const float* __restrict__ w1, const float* __restrict__ b1,
                      const float* __restrict__ w2, const float* __restrict__ b2,
                      const float* __restrict__ g2, const float* __restrict__ be2,
                      float* __restrict__ xv) {
    __shared__ float hh[D_];
    __shared__ float f[H_];
    __shared__ float red[256];
    int b = blockIdx.x, d = threadIdx.x;
    hh[d] = h1[b * D_ + d];
    __syncthreads();

    #pragma unroll
    for (int q = 0; q < 2; ++q) {
        int hc = d + q * D_;
        float a = b1[hc];
        for (int k = 0; k < D_; ++k) a += hh[k] * w1[k * H_ + hc];
        f[hc] = gelu_exact(a);
    }
    __syncthreads();

    float a = b2[d];
    for (int k = 0; k < H_; ++k) a += f[k] * w2[k * D_ + d];
    float y = x_mid[b * D_ + d] + a;

    float mean = block_sum256(y, red) * (1.0f / D_);
    float c = y - mean;
    float var = block_sum256(c * c, red) * (1.0f / D_);
    float rs = rsqrtf(var + 1e-5f);
    xv[b * D_ + d] = c * rs * g2[d] + be2[d];
}

// ---------------------------------------------------------------- head: LN + [D,V] matmul
__global__ void k_head(const float* __restrict__ xv,
                       const float* __restrict__ g, const float* __restrict__ be,
                       const float* __restrict__ hw, const float* __restrict__ hb,
                       float* __restrict__ out, int t) {
    __shared__ float xn[D_];
    __shared__ float red[256];
    int b = blockIdx.x, d = threadIdx.x;
    float v = xv[b * D_ + d];
    float mean = block_sum256(v, red) * (1.0f / D_);
    float c = v - mean;
    float var = block_sum256(c * c, red) * (1.0f / D_);
    float rs = rsqrtf(var + 1e-5f);
    xn[d] = c * rs * g[d] + be[d];
    __syncthreads();
    if (d < V_) {
        float a = hb[d];
        for (int k = 0; k < D_; ++k) a += xn[k] * hw[k * V_ + d];
        out[(b * T_ + t) * V_ + d] = a;
    }
}

// ---------------------------------------------------------------- host driver
extern "C" void kernel_launch(void* const* d_in, const int* in_sizes, int n_in,
                              void* d_out, int out_size, void* d_ws, size_t ws_size,
                              hipStream_t stream) {
    const int*   x      = (const int*)  d_in[0];
    const float* emb    = (const float*)d_in[1];
    const float* pos    = (const float*)d_in[2];
    const float* rw1    = (const float*)d_in[3];
    const float* rb1    = (const float*)d_in[4];
    const float* rw2    = (const float*)d_in[5];
    const float* rb2    = (const float*)d_in[6];
    const float* wih    = (const float*)d_in[7];
    const float* whh    = (const float*)d_in[8];
    const float* bih    = (const float*)d_in[9];
    const float* bhh    = (const float*)d_in[10];
    const float* pw     = (const float*)d_in[11];
    const float* pb     = (const float*)d_in[12];
    const float* l1g    = (const float*)d_in[13];
    const float* l1b    = (const float*)d_in[14];
    const float* fw1    = (const float*)d_in[15];
    const float* fb1    = (const float*)d_in[16];
    const float* fw2    = (const float*)d_in[17];
    const float* fb2    = (const float*)d_in[18];
    const float* l2g    = (const float*)d_in[19];
    const float* l2b    = (const float*)d_in[20];
    const float* log_   = (const float*)d_in[21];
    const float* lob    = (const float*)d_in[22];
    const float* hw     = (const float*)d_in[23];
    const float* hb     = (const float*)d_in[24];
    float* out = (float*)d_out;

    // ---- workspace carve (256B aligned) ----
    uintptr_t p = ((uintptr_t)d_ws + 255) & ~(uintptr_t)255;
    auto carve = [&](size_t bytes) {
        void* r = (void*)p;
        p += (bytes + 255) & ~(size_t)255;
        return r;
    };
    const size_t Selems = (size_t)L_ * B_ * NS_ * D_;          // 786,432
    float*    Sb0   = (float*)   carve(Selems * 4);
    float*    Sb1   = (float*)   carve(Selems * 4);
    _Float16* whh16 = (_Float16*)carve((size_t)L_ * D3_ * D_ * 2);
    float*    xv    = (float*)   carve((size_t)B_ * D_ * 4);
    float*    alpha = (float*)   carve((size_t)B_ * NS_ * 4);
    float*    Xw    = (float*)   carve((size_t)B_ * D3_ * 4);
    float*    x_mid = (float*)   carve((size_t)B_ * D_ * 4);
    float*    h1    = (float*)   carve((size_t)B_ * D_ * 4);

    // ---- per-launch setup: zero S state, convert Whh to f16 ----
    k_zero<<<(int)(Selems / 256), 256, 0, stream>>>(Sb0, (int)Selems);
    const int nwhh = L_ * D3_ * D_;
    k_cvt_f16<<<nwhh / 256, 256, 0, stream>>>(whh, whh16, nwhh);

    for (int t = 0; t < T_; ++t) {
        float* Scur = (t & 1) ? Sb1 : Sb0;
        float* Snxt = (t & 1) ? Sb0 : Sb1;
        k_embed<<<(B_ * D_) / 256, 256, 0, stream>>>(x, emb, pos, xv, t);

        for (int i = 0; i < L_; ++i) {
            const size_t lo = (size_t)i * B_ * NS_ * D_;
            k_route<<<B_, 256, 0, stream>>>(Scur + lo, xv,
                                            rw1 + (size_t)i * 2 * D_ * D_, rb1 + (size_t)i * D_,
                                            rw2 + (size_t)i * D_ * NS_,    rb2 + (size_t)i * NS_,
                                            alpha);
            k_gx<<<B_, 256, 0, stream>>>(xv, wih + (size_t)i * D3_ * D_, Xw);
            k_gru<<<192, 128, 0, stream>>>(Scur + lo, Snxt + lo,
                                           whh16 + (size_t)i * D3_ * D_,
                                           alpha, Xw,
                                           bih + (size_t)i * D3_, bhh + (size_t)i * D3_);
            k_slotproj<<<B_, 256, 0, stream>>>(alpha, Snxt + lo, xv,
                                               pw + (size_t)i * D_ * D_, pb + (size_t)i * D_,
                                               l1g + (size_t)i * D_,     l1b + (size_t)i * D_,
                                               x_mid, h1);
            k_ffn<<<B_, 256, 0, stream>>>(x_mid, h1,
                                          fw1 + (size_t)i * D_ * H_, fb1 + (size_t)i * H_,
                                          fw2 + (size_t)i * H_ * D_, fb2 + (size_t)i * D_,
                                          l2g + (size_t)i * D_,      l2b + (size_t)i * D_,
                                          xv);
        }
        k_head<<<B_, 256, 0, stream>>>(xv, log_, lob, hw, hb, out, t);
    }
}
